// Transformers_27015344292108
// MI455X (gfx1250) — matrix operand order repaired and checked by reading, not compiled, not run on hardware
//
#include <hip/hip_runtime.h>
#include <hip/hip_bf16.h>

// ---------------------------------------------------------------------------
// Types for gfx1250 WMMA (wave32): D(16x16 f32) = A(16x32 bf16) x B(32x16 bf16) + C
// ---------------------------------------------------------------------------
typedef __attribute__((ext_vector_type(16))) __bf16 v16bf;
typedef __attribute__((ext_vector_type(2)))  __bf16 v2bf;
typedef __attribute__((ext_vector_type(8)))  float  v8f;

union FragAB { v16bf v; unsigned int w[8]; unsigned short u[16]; };
union FragC  { v8f   v; float f[8]; };

// Pack two f32 -> two bf16 in one dword. Building a <2 x bf16> vector from two
// fptruncs lets the backend fuse into a single v_cvt_pk_bf16_f32.
__device__ __forceinline__ unsigned int pk2bf(float a, float b) {
#if __has_builtin(__builtin_amdgcn_cvt_pk_bf16_f32)
    union { v2bf v; unsigned int u; } c;
    c.v = __builtin_amdgcn_cvt_pk_bf16_f32(a, b);
    return c.u;
#else
    union { v2bf v; unsigned int u; } c;
    v2bf h = { (__bf16)a, (__bf16)b };
    c.v = h;
    return c.u;
#endif
}

__device__ __forceinline__ unsigned short bf1(float a) {
    union { __bf16 h[2]; unsigned short s[2]; } c;
    c.h[0] = (__bf16)a;
    return c.s[0];
}

// ---------------------------------------------------------------------------
// Tiled WMMA GEMM:  C[z] = alpha * (A[z] @ B[z]) + bias  (optional relu)
//   A: [M,K] f32 row-major (lda)
//   B: bT==0 -> [K,N] f32 row-major (weights)
//      bT==1 -> [N,K] f32 row-major (acts as B^T, e.g. Q@K^T)
//   Block = 128 threads (4 waves), tile 64x64, K-step 64 (2 WMMA k-subtiles).
//   Each wave owns a 32x32 quadrant = 2x2 WMMA tiles -> 8 wmma per K-step.
//   Double-buffered LDS: global loads for step i+1 issue during WMMAs of step i.
// ---------------------------------------------------------------------------
#define TM 64
#define TN 64
#define TK 64
#define LDK 72   // padded LDS row stride in ushorts (144B -> conflict-free row map)

__global__ __launch_bounds__(128)
void gemm_wmma_bf16(const float* __restrict__ A, int lda, long long strideA,
                    const float* __restrict__ B, int ldb, long long strideB,
                    float* __restrict__ C, int ldc, long long strideC,
                    const float* __restrict__ bias,
                    int M, int N, int K, int bT, int relu, float alpha)
{
    __shared__ unsigned short As[2][TM * LDK];
    __shared__ unsigned short Bs[2][TN * LDK];   // staged TRANSPOSED: Bs[n][k]

    const int z = blockIdx.z;
    A += (long long)z * strideA;
    B += (long long)z * strideB;
    C += (long long)z * strideC;

    const int m0   = blockIdx.y * TM;
    const int n0   = blockIdx.x * TN;
    const int tid  = threadIdx.x;
    const int lane = tid & 31;
    const int wave = tid >> 5;
    const int mw   = (wave >> 1) * 32;       // wave quadrant origin inside tile
    const int nw   = (wave & 1)  * 32;
    const int half = lane >> 4;              // 0: lanes 0-15, 1: lanes 16-31
    const int l16  = lane & 15;

    FragC acc[2][2];
#pragma unroll
    for (int t = 0; t < 2; ++t)
#pragma unroll
        for (int u = 0; u < 2; ++u)
#pragma unroll
            for (int e = 0; e < 8; ++e) acc[t][u].f[e] = 0.0f;

    // Per-thread staging coordinates (64x64 f32 tile = 1024 float4, 8 per thread).
    // A (and bT-B): slot f -> row = f>>4, kc = (f&15)*4
    // weight-B:     slot f -> kr  = f>>4, nc = (f&15)*4
    float4 ra[8], rb[8];

    auto loadT = [&](int k0) {
#pragma unroll
        for (int it = 0; it < 8; ++it) {
            int f   = tid + it * 128;
            int row = f >> 4;
            int kc  = (f & 15) << 2;
            ra[it] = *reinterpret_cast<const float4*>(
                A + (long long)(m0 + row) * lda + k0 + kc);
        }
        if (bT) {
#pragma unroll
            for (int it = 0; it < 8; ++it) {
                int f   = tid + it * 128;
                int row = f >> 4;
                int kc  = (f & 15) << 2;
                rb[it] = *reinterpret_cast<const float4*>(
                    B + (long long)(n0 + row) * ldb + k0 + kc);
            }
        } else {
#pragma unroll
            for (int it = 0; it < 8; ++it) {
                int f  = tid + it * 128;
                int kr = f >> 4;
                int nc = (f & 15) << 2;
                rb[it] = *reinterpret_cast<const float4*>(
                    B + (long long)(k0 + kr) * ldb + n0 + nc);
            }
        }
    };

    auto storeT = [&](int buf) {
#pragma unroll
        for (int it = 0; it < 8; ++it) {
            int f   = tid + it * 128;
            int row = f >> 4;
            int kc  = (f & 15) << 2;
            unsigned int* d = reinterpret_cast<unsigned int*>(&As[buf][row * LDK + kc]);
            d[0] = pk2bf(ra[it].x, ra[it].y);
            d[1] = pk2bf(ra[it].z, ra[it].w);
        }
        if (bT) {
#pragma unroll
            for (int it = 0; it < 8; ++it) {
                int f   = tid + it * 128;
                int row = f >> 4;
                int kc  = (f & 15) << 2;
                unsigned int* d = reinterpret_cast<unsigned int*>(&Bs[buf][row * LDK + kc]);
                d[0] = pk2bf(rb[it].x, rb[it].y);
                d[1] = pk2bf(rb[it].z, rb[it].w);
            }
        } else {
#pragma unroll
            for (int it = 0; it < 8; ++it) {
                int f  = tid + it * 128;
                int kr = f >> 4;
                int nc = (f & 15) << 2;
                Bs[buf][(nc + 0) * LDK + kr] = bf1(rb[it].x);
                Bs[buf][(nc + 1) * LDK + kr] = bf1(rb[it].y);
                Bs[buf][(nc + 2) * LDK + kr] = bf1(rb[it].z);
                Bs[buf][(nc + 3) * LDK + kr] = bf1(rb[it].w);
            }
        }
    };

    auto computeT = [&](int buf) {
        // 2 k-subtiles of 32, A/B fragments in contiguous k order (k = half*16 + i)
#pragma unroll
        for (int kk = 0; kk < TK; kk += 32) {
            FragAB a[2], b[2];
#pragma unroll
            for (int t = 0; t < 2; ++t) {
                // A 16x32: lane row m=l16; elem i -> k = half*16 + i
                const unsigned short* s = &As[buf][(mw + t * 16 + l16) * LDK + kk];
#pragma unroll
                for (int p = 0; p < 8; ++p) {
                    int k = (half << 4) + (p << 1);
                    a[t].w[p] = *reinterpret_cast<const unsigned int*>(s + k);
                }
            }
#pragma unroll
            for (int u = 0; u < 2; ++u) {
                // B 32x16: lane col n=l16; elem i -> k = half*16 + i
                const unsigned short* s = &Bs[buf][(nw + u * 16 + l16) * LDK + kk + (half << 4)];
#pragma unroll
                for (int p = 0; p < 8; ++p)
                    b[u].w[p] = *reinterpret_cast<const unsigned int*>(s + 2 * p);
            }
#pragma unroll
            for (int t = 0; t < 2; ++t)
#pragma unroll
                for (int u = 0; u < 2; ++u)
                    acc[t][u].v = __builtin_amdgcn_wmma_f32_16x16x32_bf16(
                        false, a[t].v, false, b[u].v, (short)0, acc[t][u].v, false, false);
        }
    };

    // ---- software-pipelined main loop (double-buffered LDS) ----
    loadT(0);
    int buf = 0;
    for (int k0 = 0; k0 < K; k0 += TK) {
        storeT(buf);                 // waits on in-flight loads, converts, ds-stores
        __syncthreads();             // tile visible to all waves
        if (k0 + TK < K) loadT(k0 + TK);   // prefetch next tile (no wait yet)
        computeT(buf);               // 8 WMMAs overlap the prefetch latency
        buf ^= 1;
    }

    // ---- epilogue: C layout: elem e -> m = half*8 + e, n = l16 ----
#pragma unroll
    for (int t = 0; t < 2; ++t) {
#pragma unroll
        for (int u = 0; u < 2; ++u) {
            int n = n0 + nw + u * 16 + l16;
            float bv = bias ? bias[n] : 0.0f;
#pragma unroll
            for (int e = 0; e < 8; ++e) {
                int m = m0 + mw + t * 16 + (half << 3) + e;
                float vv = acc[t][u].f[e] * alpha + bv;
                if (relu) vv = fmaxf(vv, 0.0f);
                C[(long long)m * ldc + n] = vv;
            }
        }
    }
}

// ---------------------------------------------------------------------------
// Embedding + sinusoidal positional encoding.  grid=(B*S), block=512 (=D)
// ---------------------------------------------------------------------------
__global__ void embed_pe(const int* __restrict__ tok, const float* __restrict__ emb,
                         float* __restrict__ out)
{
    int row = blockIdx.x;                 // b*512 + s
    int s   = row & 511;
    int d   = threadIdx.x;                // 0..511
    int t   = tok[row];
    float e = emb[(long long)t * 512 + d];
    int   i = d >> 1;
    float ang = (float)s * powf(10000.0f, -2.0f * (float)i / 256.0f);
    float pe  = (d & 1) ? cosf(ang) : sinf(ang);
    out[(long long)row * 512 + d] = e + pe;
}

// ---------------------------------------------------------------------------
// out = LayerNorm(yv + xv) * g + bta      grid=(B*S), block=256, D=512
// ---------------------------------------------------------------------------
__global__ void ln_res(const float* __restrict__ yv, const float* __restrict__ xv,
                       const float* __restrict__ g, const float* __restrict__ bta,
                       float* __restrict__ out)
{
    __shared__ float red[256];
    int row = blockIdx.x, t = threadIdx.x;
    const float* yr = yv + (long long)row * 512;
    const float* xr = xv + (long long)row * 512;
    float v0 = yr[t] + xr[t];
    float v1 = yr[t + 256] + xr[t + 256];

    red[t] = v0 + v1; __syncthreads();
    for (int o = 128; o > 0; o >>= 1) { if (t < o) red[t] += red[t + o]; __syncthreads(); }
    float mu = red[0] * (1.0f / 512.0f); __syncthreads();

    float d0 = v0 - mu, d1 = v1 - mu;
    red[t] = d0 * d0 + d1 * d1; __syncthreads();
    for (int o = 128; o > 0; o >>= 1) { if (t < o) red[t] += red[t + o]; __syncthreads(); }
    float rs = rsqrtf(red[0] * (1.0f / 512.0f) + 1e-3f);

    out[(long long)row * 512 + t]       = d0 * rs * g[t]       + bta[t];
    out[(long long)row * 512 + t + 256] = d1 * rs * g[t + 256] + bta[t + 256];
}

// ---------------------------------------------------------------------------
// Row softmax over S=512 scores with scale + optional causal mask. In-place.
// grid=(B*S) rows, block=256 (2 elems/thread).
// ---------------------------------------------------------------------------
__global__ void softmax_attn(float* __restrict__ sc, int causal, float scale)
{
    __shared__ float red[256];
    int row = blockIdx.x;                 // b*512 + q
    int q   = row & 511;
    int t   = threadIdx.x;
    float* r = sc + (long long)row * 512;
    float v0 = r[t]       * scale + ((causal && (t)       > q) ? -1e9f : 0.0f);
    float v1 = r[t + 256] * scale + ((causal && (t + 256) > q) ? -1e9f : 0.0f);

    red[t] = fmaxf(v0, v1); __syncthreads();
    for (int o = 128; o > 0; o >>= 1) { if (t < o) red[t] = fmaxf(red[t], red[t + o]); __syncthreads(); }
    float mx = red[0]; __syncthreads();

    float e0 = __expf(v0 - mx), e1 = __expf(v1 - mx);
    red[t] = e0 + e1; __syncthreads();
    for (int o = 128; o > 0; o >>= 1) { if (t < o) red[t] += red[t + o]; __syncthreads(); }
    float inv = 1.0f / red[0];
    r[t] = e0 * inv; r[t + 256] = e1 * inv;
}

// ---------------------------------------------------------------------------
// Fold Wo [H*D, D] -> Weff [D, D]: sum of the H=4 row-blocks (concat collapse).
// ---------------------------------------------------------------------------
__global__ void fold_wo(const float* __restrict__ wo, float* __restrict__ we)
{
    int i = blockIdx.x * 256 + threadIdx.x;          // 0 .. 512*512-1
    we[i] = wo[i] + wo[i + 262144] + wo[i + 2 * 262144] + wo[i + 3 * 262144];
}

// ---------------------------------------------------------------------------
// In-place softmax over VOCAB=32000.  grid=(B*S), block=512, stride loop.
// ---------------------------------------------------------------------------
__global__ void softmax_vocab(float* __restrict__ out)
{
    __shared__ float red[512];
    float* r = out + (long long)blockIdx.x * 32000;
    int t = threadIdx.x;

    float mx = -3.0e38f;
    for (int i = t; i < 32000; i += 512) mx = fmaxf(mx, r[i]);
    red[t] = mx; __syncthreads();
    for (int o = 256; o > 0; o >>= 1) { if (t < o) red[t] = fmaxf(red[t], red[t + o]); __syncthreads(); }
    mx = red[0]; __syncthreads();

    float s = 0.0f;
    for (int i = t; i < 32000; i += 512) s += __expf(r[i] - mx);
    red[t] = s; __syncthreads();
    for (int o = 256; o > 0; o >>= 1) { if (t < o) red[t] += red[t + o]; __syncthreads(); }
    float inv = 1.0f / red[0];
    for (int i = t; i < 32000; i += 512) r[i] = __expf(r[i] - mx) * inv;
}

// ---------------------------------------------------------------------------
// Launcher
// ---------------------------------------------------------------------------
extern "C" void kernel_launch(void* const* d_in, const int* in_sizes, int n_in,
                              void* d_out, int out_size, void* d_ws, size_t ws_size,
                              hipStream_t stream)
{
    (void)in_sizes; (void)n_in; (void)out_size; (void)ws_size;

    const int B = 4, S = 512, D = 512;
    const long long NX = (long long)B * S * D;       // 1,048,576 floats
    const float ISQ = 0.04419417382415922f;          // 1/sqrt(512)

    auto W = [&](int i) { return (const float*)d_in[i]; };
    // Input index map (depth-first flatten of setup_inputs()):
    // 0 vocab, 1 src, 2 tgt, 3 emb_enc, 4 emb_dec
    // enc layer l (base 5+14l): 0 qW 1 qb 2 kW 3 kb 4 vW 5 vb 6 oW 7 ob
    //                            8 nW 9 nb 10 oW' 11 ob' 12 g 13 bta
    // dec layer l (base 33+22l): +0..7 mmha, +8..15 mha, +16..19 ffn, +20 g, +21 bta
    // 77 out.W [512,32000], 78 out.b

    float* ws   = (float*)d_ws;
    float* x    = ws;              // encoder stream / enc_out
    float* y    = x    + NX;       // decoder stream
    float* qb   = y    + NX;
    float* kb   = qb   + NX;       // v == k everywhere (reference quirk)
    float* sc   = kb   + NX;       // scores [B,S,S] = NX floats
    float* att  = sc   + NX;
    float* tb   = att  + NX;
    float* r2   = tb   + NX;
    float* r3   = r2   + NX;
    float* we   = r3   + NX;       // 512*512
    float* hbuf = we   + 262144;   // 2048*2048 ffn hidden

    auto gemm = [&](const float* A, int lda, long long sA,
                    const float* Bm, int ldb, long long sB,
                    float* C, int ldc, long long sC,
                    const float* bias, int M, int N, int K,
                    int batch, int bT, int relu, float alpha) {
        dim3 g(N / 64, M / 64, batch), blk(128);
        gemm_wmma_bf16<<<g, blk, 0, stream>>>(A, lda, sA, Bm, ldb, sB, C, ldc, sC,
                                              bias, M, N, K, bT, relu, alpha);
    };

    // attention block: out = concat-collapsed output projection of softmax(QK^T/sqrt(D)) @ V
    auto attention = [&](const float* inQ, const float* inKV, int wb, int causal, float* outp) {
        gemm(inQ, 512, 0, W(wb + 0), 512, 0, qb, 512, 0, W(wb + 1), 2048, 512, 512, 1, 0, 0, 1.0f);
        gemm(inKV, 512, 0, W(wb + 2), 512, 0, kb, 512, 0, W(wb + 3), 2048, 512, 512, 1, 0, 0, 1.0f);
        // scores[b] = Q_b @ K_b^T   (batched, bT: K is [S,D] row-major)
        gemm(qb, 512, (long long)S * D, kb, 512, (long long)S * D,
             sc, 512, (long long)S * S, nullptr, 512, 512, 512, B, 1, 0, 1.0f);
        softmax_attn<<<B * S, 256, 0, stream>>>(sc, causal, ISQ);
        // att[b] = P_b @ V_b   (V == K buffer)
        gemm(sc, 512, (long long)S * S, kb, 512, (long long)S * D,
             att, 512, (long long)S * D, nullptr, 512, 512, 512, B, 0, 0, 1.0f);
        fold_wo<<<1024, 256, 0, stream>>>(W(wb + 6), we);
        gemm(att, 512, 0, we, 512, 0, outp, 512, 0, W(wb + 7), 2048, 512, 512, 1, 0, 0, 1.0f);
    };

    auto ffn = [&](const float* in, int nW, int nB, int oW, int oB, float* outp) {
        gemm(in, 512, 0, W(nW), 2048, 0, hbuf, 2048, 0, W(nB), 2048, 2048, 512, 1, 0, 1, 1.0f);
        gemm(hbuf, 2048, 0, W(oW), 512, 0, outp, 512, 0, W(oB), 2048, 512, 2048, 1, 0, 0, 1.0f);
    };

    // ---------------- encoder ----------------
    embed_pe<<<B * S, 512, 0, stream>>>((const int*)d_in[1], W(3), x);
    for (int l = 0; l < 2; ++l) {
        int base = 5 + 14 * l;
        attention(x, x, base, /*causal=*/0, tb);
        ln_res<<<B * S, 256, 0, stream>>>(tb, x, W(base + 12), W(base + 13), r2);
        ffn(r2, base + 8, base + 9, base + 10, base + 11, tb);
        ln_res<<<B * S, 256, 0, stream>>>(tb, r2, W(base + 12), W(base + 13), x);
    }
    // x == enc_out

    // ---------------- decoder ----------------
    embed_pe<<<B * S, 512, 0, stream>>>((const int*)d_in[2], W(4), y);
    for (int l = 0; l < 2; ++l) {
        int base = 33 + 22 * l;
        attention(y, y, base + 0, /*causal=*/1, tb);                 // masked self-attn
        ln_res<<<B * S, 256, 0, stream>>>(tb, y, W(base + 20), W(base + 21), r2);
        attention(r2, x, base + 8, /*causal=*/0, tb);                // cross-attn
        ln_res<<<B * S, 256, 0, stream>>>(tb, r2, W(base + 20), W(base + 21), r3);
        ffn(r3, base + 16, base + 17, base + 18, base + 19, tb);
        ln_res<<<B * S, 256, 0, stream>>>(tb, r3, W(base + 20), W(base + 21), y);
    }

    // ---------------- output projection + softmax (in-place in d_out) ----------------
    gemm(y, 512, 0, W(77), 32000, 0, (float*)d_out, 32000, 0, W(78),
         2048, 32000, 512, 1, 0, 0, 1.0f);
    softmax_vocab<<<B * S, 512, 0, stream>>>((float*)d_out);
}
